// GCNLayer_63136019251379
// MI455X (gfx1250) — matrix operand order repaired and checked by reading, not compiled, not run on hardware
//
#include <hip/hip_runtime.h>

// ---------------------------------------------------------------------------
// GCN layer for MI455X (gfx1250, wave32):
//   h   = x @ W^T                (bf16 WMMA, f32 accumulate, h stored bf16)
//   out[r] += val[e] * h[col[e]] (L2-resident gather + global_atomic_add_f32)
// ---------------------------------------------------------------------------

typedef __attribute__((ext_vector_type(16))) __bf16 v16bf;
typedef __attribute__((ext_vector_type(8)))  float  v8f;

#define DFEAT 256   // D_IN == D_OUT == 256
#define DPAIR 128   // bf16 pairs per feature row

__device__ __forceinline__ unsigned short f2bf(float f) {
    unsigned int u = __float_as_uint(f);
    return (unsigned short)((u + 0x7fffu + ((u >> 16) & 1u)) >> 16);  // RNE
}

__device__ __forceinline__ unsigned int pack_bf16x2(float lo, float hi) {
    return (unsigned int)f2bf(lo) | ((unsigned int)f2bf(hi) << 16);
}

// ---- kernel 0: zero the output (harness poisons it) -----------------------
__global__ void zero_out_kernel(float4* __restrict__ out, int n4) {
    int i = blockIdx.x * blockDim.x + threadIdx.x;
    if (i < n4) out[i] = make_float4(0.f, 0.f, 0.f, 0.f);
}

// ---- kernel 1: convert W (row-major [DOUT][DIN] f32) -> packed bf16 -------
__global__ void convert_w_kernel(const float2* __restrict__ W2,
                                 unsigned int* __restrict__ Wb, int npairs) {
    int i = blockIdx.x * blockDim.x + threadIdx.x;
    if (i < npairs) {
        float2 w = W2[i];
        Wb[i] = pack_bf16x2(w.x, w.y);
    }
}

// ---- kernel 2: h = x @ W^T via v_wmma_f32_16x16x32_bf16 -------------------
// Block = 256 threads = 8 waves. Block owns a 16-row slab of x (staged to LDS
// as bf16 once), each wave produces two 16x16 output tiles (covers all 256
// output columns). K-loop fully unrolled: 8 steps of K=32.
__global__ __launch_bounds__(256)
void gemm_bf16_kernel(const float* __restrict__ x,
                      const unsigned int* __restrict__ Wb,   // bf16 pairs [256][128]
                      unsigned short* __restrict__ hb,       // bf16 out   [N][256]
                      int n_nodes) {
    __shared__ unsigned int xs[16 * DPAIR];   // 16 rows x 256 bf16 = 8 KB

    const int tid = threadIdx.x;
    const int M0  = blockIdx.x * 16;

    // Stage x[M0:M0+16, :] into LDS as packed bf16 (cooperative, 8 pairs/thr).
    for (int p = tid; p < 16 * DPAIR; p += 256) {
        int r = p >> 7;          // row in slab
        int c = p & 127;         // pair column
        int row = M0 + r;
        if (row >= n_nodes) row = n_nodes - 1;   // clamp (no EXEC holes later)
        const float2 v = ((const float2*)(x + (size_t)row * DFEAT))[c];
        xs[p] = pack_bf16x2(v.x, v.y);
    }
    __syncthreads();

    const int lane = tid & 31;
    const int wave = tid >> 5;
    const int l16  = lane & 15;
    const int half = lane >> 4;        // 0: lanes 0-15, 1: lanes 16-31

    union Frag { unsigned int u[8]; uint4 q[2]; v16bf v; };

    v8f acc0 = {};
    v8f acc1 = {};

    const int n0 = (wave * 2) * 16;    // first N tile for this wave
    const int n1 = n0 + 16;            // second N tile

    const uint4* Arow = (const uint4*)(xs + l16 * DPAIR);
    const uint4* B0   = (const uint4*)(Wb + (size_t)(n0 + l16) * DPAIR);
    const uint4* B1   = (const uint4*)(Wb + (size_t)(n1 + l16) * DPAIR);

#pragma unroll
    for (int k0 = 0; k0 < DFEAT; k0 += 32) {
        Frag a, b0, b1;
        // A 16x32 bf16: lane needs K = k0+half*16 .. +15 (contiguous)
        const int ap0 = (k0 / 2 + half * 8) >> 2;          // uint4 index in row
        const int ap1 = ap0 + 1;
        a.q[0] = Arow[ap0];
        a.q[1] = Arow[ap1];
        // B 32x16 bf16: lane = column n0+l16, K = k0+half*16 .. +15 (contiguous)
        const int bp = (k0 / 2 + half * 8) >> 2;           // uint4 index in row
        b0.q[0] = B0[bp];  b0.q[1] = B0[bp + 1];
        b1.q[0] = B1[bp];  b1.q[1] = B1[bp + 1];

        acc0 = __builtin_amdgcn_wmma_f32_16x16x32_bf16(
                   false, a.v, false, b0.v, (short)0, acc0, false, false);
        acc1 = __builtin_amdgcn_wmma_f32_16x16x32_bf16(
                   false, a.v, false, b1.v, (short)0, acc1, false, false);
    }

    // C/D layout: VGPR i -> row M0 + half*8 + i, col = tile + l16. Store bf16.
#pragma unroll
    for (int i = 0; i < 8; ++i) {
        const int row = M0 + half * 8 + i;
        if (row < n_nodes) {
            hb[(size_t)row * DFEAT + n0 + l16] = f2bf(acc0[i]);
            hb[(size_t)row * DFEAT + n1 + l16] = f2bf(acc1[i]);
        }
    }
}

// ---- kernel 3: edge scatter: out[row] += val * h_bf16[col] ----------------
// 32 lanes per edge, 8 features/lane (one 16B bf16 load, 8 f32 atomics).
// h (51 MB) and out (102 MB) are both L2-resident on the 192 MB L2.
__global__ __launch_bounds__(256)
void scatter_kernel(const int* __restrict__ erow,
                    const int* __restrict__ ecol,
                    const float* __restrict__ eval,
                    const unsigned short* __restrict__ hb,
                    float* __restrict__ out, int n_edges) {
    const long long g = (long long)blockIdx.x * blockDim.x + threadIdx.x;
    const int e = (int)(g >> 5);
    const int t = (int)(g & 31);
    if (e >= n_edges) return;

    const int   row = erow[e];
    const int   col = ecol[e];
    const float val = eval[e];

    const uint4 hv = *(const uint4*)(hb + (size_t)col * DFEAT + t * 8);
    float* dst = out + (size_t)row * DFEAT + t * 8;

    const unsigned int uu[4] = {hv.x, hv.y, hv.z, hv.w};
#pragma unroll
    for (int j = 0; j < 4; ++j) {
        const float lo = __uint_as_float(uu[j] << 16);          // bf16 -> f32
        const float hi = __uint_as_float(uu[j] & 0xffff0000u);
        atomicAdd(dst + 2 * j,     val * lo);                   // global_atomic_add_f32
        atomicAdd(dst + 2 * j + 1, val * hi);
    }
}

// ---------------------------------------------------------------------------
extern "C" void kernel_launch(void* const* d_in, const int* in_sizes, int n_in,
                              void* d_out, int out_size, void* d_ws, size_t ws_size,
                              hipStream_t stream) {
    const float* x    = (const float*)d_in[0];
    const float* W    = (const float*)d_in[1];
    const int*   erow = (const int*)d_in[2];
    const int*   ecol = (const int*)d_in[3];
    const float* eval = (const float*)d_in[4];
    float* out = (float*)d_out;

    const int n_nodes = in_sizes[0] / DFEAT;
    const int n_edges = in_sizes[2];

    // Workspace layout: [ h bf16 : n_nodes*256*2 B ][ W bf16 : 128 KB ]
    size_t h_bytes = (size_t)n_nodes * DFEAT * sizeof(unsigned short);
    h_bytes = (h_bytes + 255) & ~(size_t)255;                 // 256B align
    unsigned short* hb = (unsigned short*)d_ws;
    unsigned int*   Wb = (unsigned int*)((char*)d_ws + h_bytes);

    // 1. convert W -> bf16 (65536 f32 -> 32768 packed pairs)
    const int wpairs = in_sizes[1] / 2;
    convert_w_kernel<<<(wpairs + 255) / 256, 256, 0, stream>>>(
        (const float2*)W, Wb, wpairs);

    // 2. zero the output accumulator
    const int n4 = out_size / 4;
    zero_out_kernel<<<(n4 + 255) / 256, 256, 0, stream>>>((float4*)out, n4);

    // 3. projection GEMM: one 16-row slab per block
    const int mblocks = (n_nodes + 15) / 16;
    gemm_bf16_kernel<<<mblocks, 256, 0, stream>>>(x, Wb, hb, n_nodes);

    // 4. edge scatter: 32 threads per edge
    const long long sthreads = (long long)n_edges * 32;
    const int sblocks = (int)((sthreads + 255) / 256);
    scatter_kernel<<<sblocks, 256, 0, stream>>>(erow, ecol, eval, hb, out, n_edges);
}
